// FC_BiLSTM_CRF_7292854469126
// MI455X (gfx1250) — compile-verified, matrix operand order checked by reading, not run on hardware
//
#include <hip/hip_runtime.h>
#include <hip/hip_bf16.h>

// ---------------------------------------------------------------------------
// Types for CDNA5 WMMA (wave32): 16x16x32 f16 -> f32
// ---------------------------------------------------------------------------
typedef _Float16 v16h __attribute__((ext_vector_type(16)));
typedef float    v8f  __attribute__((ext_vector_type(8)));
typedef _Float16 h8v  __attribute__((ext_vector_type(8)));
typedef float    f4v  __attribute__((ext_vector_type(4)));
typedef unsigned u32x4 __attribute__((ext_vector_type(4)));
typedef unsigned u32x8 __attribute__((ext_vector_type(8)));

// ---------------------------------------------------------------------------
// Tensor Data Mover: DMA a 2D tile (tile0 x tile1 elements, 4B elements) of a
// row-major f32 matrix into LDS. Descriptor per CDNA5 ISA ch.8 (D# groups 0,1;
// VADDR2/3 NULL => 2D form). Rows beyond td1 / cols beyond td0 read as zero.
// Tracked with TENSORcnt.
// ---------------------------------------------------------------------------
__device__ __forceinline__ void tdm_load_2d_f32(unsigned lds_byte_off,
                                                const float* gptr,
                                                unsigned td0, unsigned td1,
                                                unsigned stride_elems)
{
    unsigned long long ga = (unsigned long long)(uintptr_t)gptr;
    u32x4 g0;
    g0[0] = 1u;                                   // count=1, user D#
    g0[1] = lds_byte_off;                         // lds_addr (bytes)
    g0[2] = (unsigned)ga;                         // global_addr[31:0]
    g0[3] = (unsigned)((ga >> 32) & 0x1FFFFFFu)   // global_addr[56:32]
            | (2u << 30);                         // type = 2 ("image")
    u32x8 g1;
    g1[0] = 2u << 16;                             // data_size=4B; mask/flags=0
    g1[1] = td0 << 16;                            // tensor_dim0[15:0]
    g1[2] = (td0 >> 16) | (td1 << 16);            // tensor_dim0[31:16] | td1 lo
    g1[3] = (td1 >> 16) | (64u << 16);            // td1 hi | tile_dim0=64
    g1[4] = 64u;                                  // tile_dim1=64, tile_dim2=0
    g1[5] = stride_elems;                         // tensor_dim0_stride[31:0]
    g1[6] = 0u;                                   // stride[47:32], dim1_stride lo
    g1[7] = 0u;
    asm volatile("tensor_load_to_lds %0, %1" :: "s"(g0), "s"(g1) : "memory");
}

// ---------------------------------------------------------------------------
// Implicit-GEMM convolution + bias + ReLU, f16 NCHW output.
// One wave computes one 16x16 tile of Out[M=T*OH*OW, N=Cout] via WMMA.
// A gathered (im2col on the fly); B = OIHW weights (K-contiguous rows).
// ---------------------------------------------------------------------------
template <typename TIn>
__global__ void conv_gemm(const TIn* __restrict__ X, const float* __restrict__ Wt,
                          const float* __restrict__ bias, _Float16* __restrict__ Y,
                          int Cin, int H, int W, int Cout,
                          int KH, int KW, int S, int P, int OH, int OW, int K)
{
    const int lane = threadIdx.x;        // 0..31
    const int l15  = lane & 15;
    const int hi   = lane >> 4;          // 0 or 1
    const int m0   = blockIdx.x * 16;
    const int n0   = blockIdx.y * 16;
    const int OHW  = OH * OW;
    const int KHW  = KH * KW;

    const int mRow = m0 + l15;
    const int t    = mRow / OHW;
    const int rm   = mRow - t * OHW;
    const int oh   = rm / OW;
    const int ow   = rm - oh * OW;
    const int ihB  = oh * S - P;
    const int iwB  = ow * S - P;
    const TIn* Xt  = X + (size_t)t * Cin * H * W;
    const float* Brow = Wt + (size_t)(n0 + l15) * K;

    v8f acc = {};
    for (int kc = 0; kc < K; kc += 32) {
        v16h a, b;
#pragma unroll
        for (int h = 0; h < 16; ++h) {
            const int k = kc + h + (h >= 8 ? 8 : 0) + hi * 8; // ISA A-frag k map
            _Float16 av = (_Float16)0.f, bv = (_Float16)0.f;
            if (k < K) {
                const int ci = k / KHW;
                const int r2 = k - ci * KHW;
                const int kh = r2 / KW;
                const int kw = r2 - kh * KW;
                const int ih = ihB + kh, iw = iwB + kw;
                if (ih >= 0 && ih < H && iw >= 0 && iw < W)
                    av = (_Float16)(float)Xt[((size_t)ci * H + ih) * W + iw];
                bv = (_Float16)Brow[k];
            }
            a[h] = av; b[h] = bv;
        }
        acc = __builtin_amdgcn_wmma_f32_16x16x32_f16(false, a, false, b,
                                                     (short)0, acc, false, false);
    }

    const int co = n0 + l15;
    const float bv = bias[co];
#pragma unroll
    for (int r = 0; r < 8; ++r) {
        const int m  = m0 + r + hi * 8;
        const int tt = m / OHW;
        const int r3 = m - tt * OHW;
        const int o2 = r3 / OW;
        const int w2 = r3 - o2 * OW;
        float v = acc[r] + bv;
        if (v < 0.f) v = 0.f;
        Y[(((size_t)tt * Cout + co) * OH + o2) * OW + w2] = (_Float16)v;
    }
}

// ---------------------------------------------------------------------------
// TDM-fed dense GEMM: Out[M,N] = A[M,K](f16) * B[N,K]^T(f32) + bias.
// One wave per block computes a 16(M) x 64(N) tile: 4 accumulators, A fragment
// reused 4x. B panels (64 rows x 64 K) are DMA'd into LDS by the Tensor Data
// Mover, double-buffered against WMMA with s_wait_tensorcnt. B fragments are
// read from LDS (ds_load_b128) and converted f32->f16.
// Requirements: K % 64 == 0, M % 16 == 0. Nvalid guards ragged N.
// ---------------------------------------------------------------------------
__global__ void fc_gemm(const _Float16* __restrict__ A, int lda,
                        const float* __restrict__ B, int ldb,
                        const float* __restrict__ bias,
                        void* __restrict__ out, int ldc,
                        int K, int Nvalid, int relu, int outF16)
{
    __shared__ __align__(16) float Bs[2][64 * 64];   // 32 KB double buffer

    const int lane  = threadIdx.x;       // 0..31, single wave
    const int l15   = lane & 15;
    const int hi    = lane >> 4;
    const int laneK = hi * 8;
    const int m0    = blockIdx.x * 16;
    const int n0    = blockIdx.y * 64;

    const _Float16* Arow = A + (size_t)(m0 + l15) * lda;
    const float*    Bpan = B + (size_t)n0 * ldb;
    const unsigned  td1  = (unsigned)(Nvalid - n0);  // remaining B rows
    const unsigned  lds0 = (unsigned)(uintptr_t)(&Bs[0][0]); // LDS byte offset
    const unsigned  nch  = (unsigned)(K / 64);

    // Prologue: DMA chunk 0 into buffer 0.
    tdm_load_2d_f32(lds0, Bpan, (unsigned)K, td1, (unsigned)ldb);

    v8f acc[4] = {};
    for (unsigned ic = 0; ic < nch; ++ic) {
        const int kc = (int)ic * 64;
        if (ic + 1 < nch) {
            tdm_load_2d_f32(lds0 + ((ic + 1) & 1u) * (64u * 64u * 4u),
                            Bpan + kc + 64, (unsigned)(K - kc - 64), td1,
                            (unsigned)ldb);
            __builtin_amdgcn_s_wait_tensorcnt(1);    // chunk ic landed in LDS
        } else {
            __builtin_amdgcn_s_wait_tensorcnt(0);
        }
        const float* Bt = &Bs[ic & 1u][0];

#pragma unroll
        for (int kh2 = 0; kh2 < 2; ++kh2) {          // two 32-wide K halves
            const int k0 = kc + kh2 * 32 + laneK;
            const h8v alo = *(const h8v*)(Arow + k0);
            const h8v ahi = *(const h8v*)(Arow + k0 + 16);
            v16h a;
#pragma unroll
            for (int i = 0; i < 8; ++i) { a[i] = alo[i]; a[i + 8] = ahi[i]; }

            const int kl = kh2 * 32 + laneK;         // column within LDS tile
#pragma unroll
            for (int s = 0; s < 4; ++s) {            // 4 N-subtiles reuse A
                const float* Br = Bt + (s * 16 + l15) * 64;
                const f4v b0 = *(const f4v*)(Br + kl);       // ds_load_b128
                const f4v b1 = *(const f4v*)(Br + kl + 4);
                const f4v b2 = *(const f4v*)(Br + kl + 16);
                const f4v b3 = *(const f4v*)(Br + kl + 20);
                v16h b;
#pragma unroll
                for (int i = 0; i < 4; ++i) {
                    b[i]      = (_Float16)b0[i];
                    b[i + 4]  = (_Float16)b1[i];
                    b[i + 8]  = (_Float16)b2[i];
                    b[i + 12] = (_Float16)b3[i];
                }
                acc[s] = __builtin_amdgcn_wmma_f32_16x16x32_f16(
                             false, a, false, b, (short)0, acc[s], false, false);
            }
        }
    }

    // Store: n = lane&15 (+16s per subtile); m = r + 8*hi (ISA D layout).
#pragma unroll
    for (int s = 0; s < 4; ++s) {
        const int n = n0 + s * 16 + l15;
        if (n < Nvalid) {
            const float bv = bias ? bias[n] : 0.f;
#pragma unroll
            for (int r = 0; r < 8; ++r) {
                const int m = m0 + r + hi * 8;
                float v = acc[s][r] + bv;
                if (relu && v < 0.f) v = 0.f;
                if (outF16) ((_Float16*)out)[(size_t)m * ldc + n] = (_Float16)v;
                else        ((float*)out)[(size_t)m * ldc + n] = v;
            }
        }
    }
}

// ---------------------------------------------------------------------------
// 3x3 stride-2 VALID max-pool on f16 NCHW (T fixed at 128).
// ---------------------------------------------------------------------------
__global__ void maxpool_h(const _Float16* __restrict__ X, _Float16* __restrict__ Y,
                          int C, int H, int W, int OH, int OW)
{
    const int idx = blockIdx.x * blockDim.x + threadIdx.x;
    const int total = 128 * C * OH * OW;
    if (idx >= total) return;
    const int ow = idx % OW; int tmp = idx / OW;
    const int oh = tmp % OH; tmp /= OH;
    const int c  = tmp % C;  const int t = tmp / C;
    const _Float16* p = X + ((size_t)t * C + c) * H * W;
    float m = -3.4e38f;
#pragma unroll
    for (int dy = 0; dy < 3; ++dy)
#pragma unroll
        for (int dx = 0; dx < 3; ++dx) {
            const float v = (float)p[(oh * 2 + dy) * W + (ow * 2 + dx)];
            if (v > m) m = v;
        }
    Y[idx] = (_Float16)m;
}

__global__ void zero_h(_Float16* __restrict__ p, int n)
{
    const int i = blockIdx.x * blockDim.x + threadIdx.x;
    if (i < n) p[i] = (_Float16)0.f;
}

// Zero-pad rows: dst[r][0..ldDst) = src[r][0..cols) then zeros (wih 1000->1024).
__global__ void pad_rows(const float* __restrict__ src, float* __restrict__ dst,
                         int rows, int cols, int ldDst)
{
    const int idx = blockIdx.x * blockDim.x + threadIdx.x;
    if (idx >= rows * ldDst) return;
    const int r = idx / ldDst, c = idx - r * ldDst;
    dst[idx] = (c < cols) ? src[r * cols + c] : 0.f;
}

// ---------------------------------------------------------------------------
// LSTM recurrence for one direction. gpre[t][j] = (Wih x_t + bih)_j precomputed
// by WMMA GEMM. One 1024-thread workgroup: h/c/gates in LDS, 2 barriers/step.
// dir=0: forward, writes out[t][0:256); dir=1: backward, writes out[t][256:512).
// ---------------------------------------------------------------------------
__global__ void lstm_scan(const float* __restrict__ gpre, const float* __restrict__ whh,
                          const float* __restrict__ bhh, const float* __restrict__ h0,
                          const float* __restrict__ c0, _Float16* __restrict__ out, int dir)
{
    __shared__ float h_s[256];
    __shared__ float c_s[256];
    __shared__ float gates[1024];
    const int j = threadIdx.x; // 0..1023 (gate rows: i,f,g,o chunks of 256)
    if (j < 256) { h_s[j] = h0[j]; c_s[j] = c0[j]; }
    __syncthreads();
    const float* wr = whh + (size_t)j * 256;
    const float bj = bhh[j];
    for (int s = 0; s < 128; ++s) {
        const int t = dir ? (127 - s) : s;
        float acc = gpre[(size_t)t * 1024 + j] + bj;
        for (int k = 0; k < 256; ++k) acc += wr[k] * h_s[k]; // LDS broadcast reads
        gates[j] = acc;
        __syncthreads();
        if (j < 256) {
            const float ig = 1.f / (1.f + __expf(-gates[j]));
            const float fg = 1.f / (1.f + __expf(-gates[j + 256]));
            const float gg = tanhf(gates[j + 512]);
            const float og = 1.f / (1.f + __expf(-gates[j + 768]));
            const float c  = fg * c_s[j] + ig * gg;
            const float h  = og * tanhf(c);
            c_s[j] = c; h_s[j] = h;
            out[(size_t)t * 512 + dir * 256 + j] = (_Float16)h;
        }
        __syncthreads();
    }
}

// ---------------------------------------------------------------------------
// Viterbi decode, K=12, T=128. out[0]=score, out[1..128]=path (as float).
// ---------------------------------------------------------------------------
__global__ void viterbi_k(const float* __restrict__ feats,  // [128,12]
                          const float* __restrict__ trans,  // [12,12] (next,prev)
                          float* __restrict__ out)
{
    if (threadIdx.x != 0 || blockIdx.x != 0) return;
    const int K = 12, T = 128, START = 10, STOP = 11;
    float fv[12];
    int bp[128][12];
    for (int i = 0; i < K; ++i) fv[i] = -10000.f;
    fv[START] = 0.f;
    for (int t = 0; t < T; ++t) {
        float nfv[12];
        for (int n = 0; n < K; ++n) {
            float best = -3.4e38f; int bi = 0;
            for (int p = 0; p < K; ++p) {
                const float sc = fv[p] + trans[n * K + p];
                if (sc > best) { best = sc; bi = p; }
            }
            bp[t][n] = bi;
            nfv[n] = best + feats[t * K + n];
        }
        for (int n = 0; n < K; ++n) fv[n] = nfv[n];
    }
    float best = -3.4e38f; int bi = 0;
    for (int p = 0; p < K; ++p) {
        const float sc = fv[p] + trans[STOP * K + p];
        if (sc > best) { best = sc; bi = p; }
    }
    out[0] = best;
    int tag = bi;
    for (int t = T - 1; t >= 0; --t) {
        out[1 + t] = (float)tag;
        tag = bp[t][tag];
    }
}

// ---------------------------------------------------------------------------
// Host orchestration. Workspace layout (bytes):
//   [0,56M)   bufA (f16 activations, ping)   peak: conv1 out 49.6MB
//   [56M,72M) bufB (f16 activations, pong)   peak: pool1 12MB
//   [72M,76M) wih_f padded f32 [1024,1024]
//   [76M,80M) wih_b padded f32 [1024,1024]
//   [80M, +512K) gpre_f f32 [128,1024];  [+512K,+1M) gpre_b
//   [81M, +128K) lstm_out f16 [128,512]; [+256K,...) tag_feats f32 [128,12]
// ---------------------------------------------------------------------------
extern "C" void kernel_launch(void* const* d_in, const int* in_sizes, int n_in,
                              void* d_out, int out_size, void* d_ws, size_t ws_size,
                              hipStream_t stream)
{
    (void)in_sizes; (void)n_in; (void)out_size; (void)ws_size;
    const float* sentence = (const float*)d_in[0];
    const float* w1 = (const float*)d_in[1];   const float* b1 = (const float*)d_in[2];
    const float* w2 = (const float*)d_in[3];   const float* b2 = (const float*)d_in[4];
    const float* w3 = (const float*)d_in[5];   const float* b3 = (const float*)d_in[6];
    const float* w4 = (const float*)d_in[7];   const float* b4 = (const float*)d_in[8];
    const float* w5 = (const float*)d_in[9];   const float* b5 = (const float*)d_in[10];
    const float* fc6_w = (const float*)d_in[11]; const float* fc6_b = (const float*)d_in[12];
    const float* fc7_w = (const float*)d_in[13]; const float* fc7_b = (const float*)d_in[14];
    const float* fc8_w = (const float*)d_in[15]; const float* fc8_b = (const float*)d_in[16];
    const float* wih_f = (const float*)d_in[17]; const float* whh_f = (const float*)d_in[18];
    const float* bih_f = (const float*)d_in[19]; const float* bhh_f = (const float*)d_in[20];
    const float* wih_b = (const float*)d_in[21]; const float* whh_b = (const float*)d_in[22];
    const float* bih_b = (const float*)d_in[23]; const float* bhh_b = (const float*)d_in[24];
    const float* h0    = (const float*)d_in[25]; const float* c0    = (const float*)d_in[26];
    const float* h2t_w = (const float*)d_in[27]; const float* h2t_b = (const float*)d_in[28];
    const float* trans = (const float*)d_in[29];
    float* out = (float*)d_out;

    char* ws = (char*)d_ws;
    _Float16* bufA     = (_Float16*)(ws);
    _Float16* bufB     = (_Float16*)(ws + (56ull << 20));
    float*    wihf_pad = (float*)(ws + (72ull << 20));
    float*    wihb_pad = (float*)(ws + (76ull << 20));
    float*    gpre_f   = (float*)(ws + (80ull << 20));
    float*    gpre_b   = (float*)(ws + (80ull << 20) + (1ull << 19));
    _Float16* lstm_out = (_Float16*)(ws + (81ull << 20));
    float*    tag_feats= (float*)(ws + (81ull << 20) + (1ull << 18));

    const dim3 wv(32);

    // --- AlexNet ---
    conv_gemm<float><<<dim3((128*55*55)/16, 64/16), wv, 0, stream>>>(
        sentence, w1, b1, bufA, 3, 224, 224, 64, 11, 11, 4, 2, 55, 55, 363);
    { int tot = 128*64*27*27;
      maxpool_h<<<(tot+255)/256, 256, 0, stream>>>(bufA, bufB, 64, 55, 55, 27, 27); }
    conv_gemm<_Float16><<<dim3((128*27*27)/16, 192/16), wv, 0, stream>>>(
        bufB, w2, b2, bufA, 64, 27, 27, 192, 5, 5, 1, 2, 27, 27, 1600);
    { int tot = 128*192*13*13;
      maxpool_h<<<(tot+255)/256, 256, 0, stream>>>(bufA, bufB, 192, 27, 27, 13, 13); }
    conv_gemm<_Float16><<<dim3((128*13*13)/16, 384/16), wv, 0, stream>>>(
        bufB, w3, b3, bufA, 192, 13, 13, 384, 3, 3, 1, 1, 13, 13, 1728);
    conv_gemm<_Float16><<<dim3((128*13*13)/16, 256/16), wv, 0, stream>>>(
        bufA, w4, b4, bufB, 384, 13, 13, 256, 3, 3, 1, 1, 13, 13, 3456);
    conv_gemm<_Float16><<<dim3((128*13*13)/16, 256/16), wv, 0, stream>>>(
        bufB, w5, b5, bufA, 256, 13, 13, 256, 3, 3, 1, 1, 13, 13, 2304);
    { int tot = 128*256*6*6;
      maxpool_h<<<(tot+255)/256, 256, 0, stream>>>(bufA, bufB, 256, 13, 13, 6, 6); }

    // --- FC layers (TDM-fed WMMA GEMM; grid.y tiles N by 64) ---
    fc_gemm<<<dim3(128/16, 4096/64), wv, 0, stream>>>(bufB, 9216, fc6_w, 9216, fc6_b,
                                                      bufA, 4096, 9216, 4096, 1, 1);
    fc_gemm<<<dim3(128/16, 4096/64), wv, 0, stream>>>(bufA, 4096, fc7_w, 4096, fc7_b,
                                                      bufB, 4096, 4096, 4096, 1, 1);
    // fc8 into zero-padded feats [128,1024] (K padded for the LSTM GEMM)
    { int tot = 128*1024; zero_h<<<(tot+255)/256, 256, 0, stream>>>(bufA, tot); }
    fc_gemm<<<dim3(128/16, (1000 + 63)/64), wv, 0, stream>>>(bufB, 4096, fc8_w, 4096, fc8_b,
                                                             bufA, 1024, 4096, 1000, 0, 1);

    // --- LSTM input projections: pad Wih rows 1000->1024, then WMMA GEMMs ---
    { int tot = 1024*1024;
      pad_rows<<<(tot+255)/256, 256, 0, stream>>>(wih_f, wihf_pad, 1024, 1000, 1024);
      pad_rows<<<(tot+255)/256, 256, 0, stream>>>(wih_b, wihb_pad, 1024, 1000, 1024); }
    fc_gemm<<<dim3(128/16, 1024/64), wv, 0, stream>>>(bufA, 1024, wihf_pad, 1024, bih_f,
                                                      gpre_f, 1024, 1024, 1024, 0, 0);
    fc_gemm<<<dim3(128/16, 1024/64), wv, 0, stream>>>(bufA, 1024, wihb_pad, 1024, bih_b,
                                                      gpre_b, 1024, 1024, 1024, 0, 0);

    // --- Recurrent scans (LDS-resident hidden state) ---
    lstm_scan<<<1, 1024, 0, stream>>>(gpre_f, whh_f, bhh_f, h0,       c0,       lstm_out, 0);
    lstm_scan<<<1, 1024, 0, stream>>>(gpre_b, whh_b, bhh_b, h0 + 256, c0 + 256, lstm_out, 1);

    // --- Tag projection [128,512] x [12,512]^T ---
    fc_gemm<<<dim3(128/16, 1), wv, 0, stream>>>(lstm_out, 512, h2t_w, 512, h2t_b,
                                                tag_feats, 12, 512, 12, 0, 0);

    // --- Viterbi decode ---
    viterbi_k<<<1, 1, 0, stream>>>(tag_feats, trans, out);
}
